// mDACF_53541062312314
// MI455X (gfx1250) — compile-verified, matrix operand order checked by reading, not run on hardware
//
#include <hip/hip_runtime.h>

// mDA-CF fused towers for MI455X (gfx1250, wave32):
//  - user/item GEMM (+bias,+ReLU) via bf16 hi/lo 3-product WMMA (fp32-class accuracy),
//  - double-buffered LDS staging of W slabs (swizzled to B-fragment order),
//  - software-pipelined A loads, non-temporal x streaming,
//  - pair-granular bf16 packing via convertvector -> v_cvt_pk_bf16_f32,
//  - row-wise dot reduction kernel.

#define B_ROWS   8192
#define USER_DIM 3706
#define ITEM_DIM 6040
#define X_W      (USER_DIM + ITEM_DIM)   // 9746
#define LATENT   512

typedef __attribute__((ext_vector_type(16))) __bf16   v16bf;
typedef __attribute__((ext_vector_type(2)))  __bf16   bf2;
typedef __attribute__((ext_vector_type(8)))  float    v8f;
typedef __attribute__((ext_vector_type(8)))  unsigned v8u;
typedef __attribute__((ext_vector_type(2)))  float    f2v;

#define LDS_UINTS_PER_BUF (2 * 32 * 32 * 8)   // 16384 uints = 64 KB per slab buffer

// Split a fp32 pair into packed-bf16 hi and lo (residual) words.
// Lowers to: v_cvt_pk_bf16_f32 (hi), 2x v_fma_mix_f32_bf16 (residual),
// v_cvt_pk_bf16_f32 (lo) -- no v_mov_b16 re-packing.
__device__ __forceinline__ void split_pack_bf16(f2v v, unsigned& hi, unsigned& lo) {
    const bf2 h = __builtin_convertvector(v, bf2);
    hi = __builtin_bit_cast(unsigned, h);
    const f2v hf = __builtin_convertvector(h, f2v);
    const bf2 l  = __builtin_convertvector(v - hf, bf2);
    lo = __builtin_bit_cast(unsigned, l);
}

// Stage W[0..511][kb..kb+31] into LDS as hi/lo bf16 in B-fragment order.
// Fragment layout (16-bit B, 32x16): lane = (n&15) + 16*h, uint u = g*4 + r,
// for tile-local kk = 2j: g = kk/16, h = (kk%16)/8, r = (kk%8)/2.
// 16B chunks are XOR-swizzled by ((lane>>3)&1) so the two ds_load_b128 of a
// fragment hit all 16 bank groups exactly twice (the 2-cycle minimum).
__device__ __forceinline__ void stage_slab(unsigned* __restrict__ buf,
                                           const float* __restrict__ W,
                                           int K, int kb, int tid)
{
    #pragma unroll 4
    for (int e = tid; e < 512 * 16; e += 256) {
        const int n = e >> 4;
        const int j = e & 15;
        const int k = kb + j * 2;
        f2v v = {0.0f, 0.0f};
        if (k < K)                        // K even => pair never straddles edge
            v = *(const f2v*)(W + (size_t)n * K + k);

        unsigned hiw, low;
        split_pack_bf16(v, hiw, low);

        const int t = n >> 4;
        const int l = (n & 15) | (((j >> 2) & 1) << 4);
        const int u = ((j >> 3) << 2) | (j & 3);
        const int c = u >> 2;                         // which 16B chunk (u0..3 / u4..7)
        const int chunk = (l * 2 + c) ^ ((l >> 3) & 1);
        buf[(      t) * 256 + chunk * 4 + (u & 3)] = hiw;
        buf[(32 +  t) * 256 + chunk * 4 + (u & 3)] = low;
    }
}

// Raw fp32 A pair loads for one K-step (issued one step ahead; non-temporal:
// x is streamed exactly once and must not evict the L2-resident weights).
__device__ __forceinline__ void load_A_raw(f2v* __restrict__ av,
                                           const float* __restrict__ xrow,
                                           int kb, int half, int K)
{
    #pragma unroll
    for (int q = 0; q < 4; ++q) {
        const int k0 = kb + half * 8 + q * 2;          // K group 0
        const int k1 = kb + 16 + half * 8 + q * 2;     // K group 1
        f2v v0 = {0.0f, 0.0f}, v1 = {0.0f, 0.0f};
        if (k0 < K) v0 = __builtin_nontemporal_load((const f2v*)(xrow + k0));
        if (k1 < K) v1 = __builtin_nontemporal_load((const f2v*)(xrow + k1));
        av[q]     = v0;
        av[4 + q] = v1;
    }
}

// ---------------------------------------------------------------------------
// Kernel 1: one tower GEMM per blockIdx.y.
//   out[b, n] = relu( sum_k x[b, colOff+k] * W[n, k] + bias[n] )
// WG = 64 rows x 512 latent, 8 waves; wave w -> row-tile (w&3), latent-half
// (w>>2) = 16 accum tiles (128 VGPRs). acc += Ah*Bh + Ah*Bl + Al*Bh.
// ---------------------------------------------------------------------------
__global__ __launch_bounds__(256, 1)
void mdacf_tower_kernel(const float* __restrict__ x,
                        const float* __restrict__ Wu, const float* __restrict__ bu,
                        const float* __restrict__ Wi, const float* __restrict__ bi,
                        float* __restrict__ ws)
{
    extern __shared__ unsigned lds[];     // 2 x 64 KB double-buffered W slab

    const int    tower  = blockIdx.y;
    const float* W      = tower ? Wi : Wu;
    const float* bias   = tower ? bi : bu;
    const int    K      = tower ? ITEM_DIM : USER_DIM;   // both even
    const int    colOff = tower ? USER_DIM : 0;
    float*       out    = ws + (size_t)tower * B_ROWS * LATENT;

    const int tid  = threadIdx.x;
    const int lane = tid & 31;
    const int wave = tid >> 5;
    const int rt   = wave & 3;        // row tile within WG (16 rows each)
    const int nh   = wave >> 2;       // latent half: [nh*256, nh*256+256)
    const int rowBase = blockIdx.x * 64;

    const int half = lane >> 4;       // lane half -> K sub-range of A fragment
    const int mlow = lane & 15;

    v8f acc[16];
    #pragma unroll
    for (int t = 0; t < 16; ++t)
        #pragma unroll
        for (int j = 0; j < 8; ++j) acc[t][j] = 0.0f;

    const float* xrow = x + (size_t)(rowBase + rt * 16 + mlow) * X_W + colOff;

    unsigned* buf0 = lds;
    unsigned* buf1 = lds + LDS_UINTS_PER_BUF;

    // Swizzled fragment-chunk offsets for this lane (16B aligned).
    const int swz  = (lane >> 3) & 1;
    const int off0 = (((lane * 2)     ^ swz) << 2);   // uints u0..3
    const int off1 = (((lane * 2 + 1) ^ swz) << 2);   // uints u4..7

    const int nsteps = (K + 31) / 32;

    // Pipeline prologue: stage slab 0, load A raws for step 0.
    f2v av[8];
    stage_slab(buf0, W, K, 0, tid);
    load_A_raw(av, xrow, 0, half, K);

    for (int ks = 0; ks < nsteps; ++ks) {
        __syncthreads();   // slab[ks&1] staging complete; slab[(ks+1)&1] free

        // Convert this step's A pairs into hi/lo bf16 fragments.
        v8u rah, ral;
        #pragma unroll
        for (int p = 0; p < 8; ++p) {
            unsigned hiw, low;
            split_pack_bf16(av[p], hiw, low);
            rah[p] = hiw;
            ral[p] = low;
        }
        const v16bf Ahi = __builtin_bit_cast(v16bf, rah);
        const v16bf Alo = __builtin_bit_cast(v16bf, ral);

        // Prefetch next slab + next A raws; overlaps with this step's WMMAs.
        if (ks + 1 < nsteps) {
            unsigned* nbuf = ((ks + 1) & 1) ? buf1 : buf0;
            stage_slab(nbuf, W, K, (ks + 1) * 32, tid);
            load_A_raw(av, xrow, (ks + 1) * 32, half, K);
        }

        const unsigned* buf = (ks & 1) ? buf1 : buf0;

        // 16 N-tiles x 3 WMMAs (bf16x3 split).
        #pragma unroll
        for (int tt = 0; tt < 16; ++tt) {
            const int tg   = nh * 16 + tt;
            const int bhi  = tg * 256;
            const int blo  = (32 + tg) * 256;
            const uint4 h0 = *(const uint4*)&buf[bhi + off0];
            const uint4 h1 = *(const uint4*)&buf[bhi + off1];
            const uint4 l0 = *(const uint4*)&buf[blo + off0];
            const uint4 l1 = *(const uint4*)&buf[blo + off1];
            v8u rbh, rbl;
            rbh[0] = h0.x; rbh[1] = h0.y; rbh[2] = h0.z; rbh[3] = h0.w;
            rbh[4] = h1.x; rbh[5] = h1.y; rbh[6] = h1.z; rbh[7] = h1.w;
            rbl[0] = l0.x; rbl[1] = l0.y; rbl[2] = l0.z; rbl[3] = l0.w;
            rbl[4] = l1.x; rbl[5] = l1.y; rbl[6] = l1.z; rbl[7] = l1.w;
            const v16bf Bhi = __builtin_bit_cast(v16bf, rbh);
            const v16bf Blo = __builtin_bit_cast(v16bf, rbl);

            acc[tt] = __builtin_amdgcn_wmma_f32_16x16x32_bf16(
                          false, Ahi, false, Bhi, (short)0, acc[tt], false, false);
            acc[tt] = __builtin_amdgcn_wmma_f32_16x16x32_bf16(
                          false, Ahi, false, Blo, (short)0, acc[tt], false, false);
            acc[tt] = __builtin_amdgcn_wmma_f32_16x16x32_bf16(
                          false, Alo, false, Bhi, (short)0, acc[tt], false, false);
        }
    }

    // Epilogue: +bias, ReLU, store embeddings.
    // C layout: lane<16: (M=v, N=lane); lane>=16: (M=v+8, N=lane-16).
    #pragma unroll
    for (int tt = 0; tt < 16; ++tt) {
        const int   n  = nh * 256 + tt * 16 + mlow;
        const float bn = bias[n];
        #pragma unroll
        for (int v = 0; v < 8; ++v) {
            const int row = rowBase + rt * 16 + v + half * 8;
            float val = acc[tt][v] + bn;
            val = val > 0.0f ? val : 0.0f;
            out[(size_t)row * LATENT + n] = val;
        }
    }
}

// ---------------------------------------------------------------------------
// Kernel 2: diag = rowwise dot of the two relu'd embeddings.
// One wave per row, shuffle reduction.
// ---------------------------------------------------------------------------
__global__ __launch_bounds__(256, 1)
void mdacf_dot_kernel(const float* __restrict__ ws, float* __restrict__ out)
{
    const float* u  = ws;
    const float* it = ws + (size_t)B_ROWS * LATENT;
    const int lane = threadIdx.x & 31;
    const int wave = threadIdx.x >> 5;
    const int row  = blockIdx.x * 8 + wave;

    const float* ur = u  + (size_t)row * LATENT;
    const float* ir = it + (size_t)row * LATENT;

    float s = 0.0f;
    #pragma unroll
    for (int i = 0; i < LATENT / 32; ++i) {
        const int idx = lane + i * 32;
        s = fmaf(ur[idx], ir[idx], s);
    }
    #pragma unroll
    for (int off = 16; off > 0; off >>= 1)
        s += __shfl_xor(s, off, 32);
    if (lane == 0) out[row] = s;
}

// ---------------------------------------------------------------------------
extern "C" void kernel_launch(void* const* d_in, const int* in_sizes, int n_in,
                              void* d_out, int out_size, void* d_ws, size_t ws_size,
                              hipStream_t stream)
{
    (void)in_sizes; (void)n_in; (void)out_size; (void)ws_size;
    const float* x  = (const float*)d_in[0];
    const float* Wu = (const float*)d_in[1];
    const float* bu = (const float*)d_in[2];
    const float* Wi = (const float*)d_in[3];
    const float* bi = (const float*)d_in[4];
    float* out = (float*)d_out;
    float* ws  = (float*)d_ws;   // needs 2 * 8192 * 512 * 4 B = 32 MB

    dim3 grid(B_ROWS / 64, 2);   // x: 64-row chunks, y: tower (0=user, 1=item)
    mdacf_tower_kernel<<<grid, 256, 2 * 64 * 1024, stream>>>(x, Wu, bu, Wi, bi, ws);
    mdacf_dot_kernel<<<B_ROWS / 8, 256, 0, stream>>>(ws, out);
}
